// AttentionalFactorizationMachine_31473520345930
// MI455X (gfx1250) — matrix operand order repaired and checked by reading, not compiled, not run on hardware
//
#include <hip/hip_runtime.h>
#include <hip/hip_bf16.h>
#include <math.h>

// AFM: B=2048, F=50 fields, D=64 factor dim, ATT=64, P=F*(F-1)/2=1225 pairs.
#define NF    50
#define CARD  10000
#define DIM   64
#define ATT   64
#define NPAIR 1225
#define NTILE 77            // ceil(1225/16)
#define PPAD  (NTILE * 16)  // 1232

typedef __attribute__((ext_vector_type(16))) _Float16 v16h;
typedef __attribute__((ext_vector_type(8)))  float    v8f;
typedef __attribute__((ext_vector_type(4)))  float    v4f;

__launch_bounds__(256)
__global__ void afm_wmma_kernel(const int*   __restrict__ x,      // [B, NF]
                                const float* __restrict__ emb,    // [NF*CARD, DIM]
                                const float* __restrict__ W1,     // [DIM, ATT]
                                const float* __restrict__ b1,     // [ATT]
                                const float* __restrict__ w2,     // [ATT]
                                const float* __restrict__ b2,     // [1]
                                const float* __restrict__ lin_w,  // [NF*CARD]
                                const float* __restrict__ lin_b,  // [1]
                                float*       __restrict__ out)    // [B]
{
    __shared__ v4f    sFacV[(NF * DIM) / 4];   // 12.8 KB  factors f32
    __shared__ v4f    sW1V[(DIM * ATT) / 4];   // 16 KB    W1 f32
    __shared__ float  sLogit[PPAD];            // per-pair logits
    __shared__ float  sISum[PPAD];             // per-pair inter_sum
    __shared__ unsigned short sPair[PPAD];     // i | (j<<8)
    __shared__ float  sRed[256];
    __shared__ float  sLinear;

    float* sFac = reinterpret_cast<float*>(sFacV);
    float* sW1  = reinterpret_cast<float*>(sW1V);

    const int b    = blockIdx.x;
    const int tid  = threadIdx.x;
    const int lane = tid & 31;
    const int wave = tid >> 5;     // 0..7

    // ---- phase 0: pair index table (clamped for pad rows) + init --------
    if (tid == 0) sLinear = 0.0f;
    for (int p = tid; p < PPAD; p += 256) {
        int pp = (p < NPAIR) ? p : (NPAIR - 1);
        int i = 0, rem = pp, cnt = NF - 1;
        while (rem >= cnt) { rem -= cnt; --cnt; ++i; }
        sPair[p] = (unsigned short)(i | ((i + 1 + rem) << 8));
    }
    __syncthreads();

    // ---- phase 1: stage factors + W1 in LDS, linear term ----------------
    for (int e = tid; e < NF * (DIM / 4); e += 256) {
        const int f  = e >> 4;       // / (DIM/4)
        const int d4 = e & 15;
        const long gid = (long)x[b * NF + f] + (long)f * CARD;
        sFacV[e] = ((const v4f*)(emb + gid * DIM))[d4];   // global_load_b128
    }
    for (int e = tid; e < (DIM * ATT) / 4; e += 256) {
        sW1V[e] = ((const v4f*)W1)[e];
    }
    if (tid < NF) {
        const long gid = (long)x[b * NF + tid] + (long)tid * CARD;
        atomicAdd(&sLinear, lin_w[gid]);      // ds_add_f32
    }
    __syncthreads();

    // ---- phase 2: per-wave register-resident B fragments of W1 ---------
    // B layout (16-bit 32x16): lane%16 = column N, lane/16 selects K octets;
    // element e of v16h -> K = kstep*32 + (lane/16)*8 + (e < 8 ? e : e + 8)
    const int m  = lane & 15;
    const int hi = lane >> 4;
    v16h Bf[2][4];
#pragma unroll
    for (int ks = 0; ks < 2; ++ks) {
#pragma unroll
        for (int nt = 0; nt < 4; ++nt) {
            const int ncol  = nt * 16 + m;
            const int kbase = ks * 32 + hi * 8;
            v16h v;
#pragma unroll
            for (int e = 0; e < 16; ++e)
                v[e] = (_Float16)sW1[(kbase + (e < 8 ? e : e + 8)) * ATT + ncol];
            Bf[ks][nt] = v;
        }
    }
    // per-lane epilogue constants for the 4 N-tiles
    float b1v[4], w2v[4];
#pragma unroll
    for (int nt = 0; nt < 4; ++nt) {
        b1v[nt] = b1[nt * 16 + m];
        w2v[nt] = w2[nt * 16 + m];
    }
    const float b2v = b2[0];

    // ---- phase 3: tiles of 16 pairs -> WMMA GEMM + fused epilogue ------
    for (int t = wave; t < NTILE; t += 8) {
        const int p = t * 16 + m;
        const unsigned int pr = sPair[p];          // ds_load_u16 (clamped table)
        const int fi = pr & 0xff;
        const int fj = pr >> 8;
        const v4f* ri = (const v4f*)(sFac + fi * DIM);
        const v4f* rj = (const v4f*)(sFac + fj * DIM);

        // A fragments (16-bit 16x32 layout):
        //   elems 0..7  -> K = ks*32 + hi*8 + e        (chunk c4, c4+1)
        //   elems 8..15 -> K = ks*32 + 16 + hi*8 + e-8 (chunk c4+4, c4+5)
        float isum = 0.0f;
        v16h Af[2];
#pragma unroll
        for (int ks = 0; ks < 2; ++ks) {
            const int c4 = ks * 8 + hi * 2;        // float4 units
            const v4f p0 = ri[c4]     * rj[c4];    // 4x ds_load_b128 per row
            const v4f p1 = ri[c4 + 1] * rj[c4 + 1];
            const v4f p2 = ri[c4 + 4] * rj[c4 + 4];
            const v4f p3 = ri[c4 + 5] * rj[c4 + 5];
            v16h a;
#pragma unroll
            for (int e = 0; e < 4; ++e) {
                a[e]      = (_Float16)p0[e];
                a[e + 4]  = (_Float16)p1[e];
                a[e + 8]  = (_Float16)p2[e];
                a[e + 12] = (_Float16)p3[e];
                isum += (p0[e] + p1[e]) + (p2[e] + p3[e]);
            }
            Af[ks] = a;
        }

        // 8x v_wmma_f32_16x16x32_f16: [16 pairs x 64] @ [64 x 64]
        v8f c[4];
#pragma unroll
        for (int nt = 0; nt < 4; ++nt) {
            v8f acc = {};
            acc = __builtin_amdgcn_wmma_f32_16x16x32_f16(
                    false, Af[0], false, Bf[0][nt], (short)0, acc, false, false);
            acc = __builtin_amdgcn_wmma_f32_16x16x32_f16(
                    false, Af[1], false, Bf[1][nt], (short)0, acc, false, false);
            c[nt] = acc;
        }

        // epilogue: h = relu(c + b1); partial logit = sum_n h * w2[n]
        float lg[8] = {0.f, 0.f, 0.f, 0.f, 0.f, 0.f, 0.f, 0.f};
#pragma unroll
        for (int nt = 0; nt < 4; ++nt) {
#pragma unroll
            for (int r = 0; r < 8; ++r) {
                float h = c[nt][r] + b1v[nt];
                h = h > 0.0f ? h : 0.0f;
                lg[r] += h * w2v[nt];
            }
        }
        // reduce over the 16-lane N group (rows M=r+hi*8 live per half-wave)
#pragma unroll
        for (int r = 0; r < 8; ++r) {
            float v = lg[r];
            v += __shfl_xor(v, 1);
            v += __shfl_xor(v, 2);
            v += __shfl_xor(v, 4);
            v += __shfl_xor(v, 8);
            lg[r] = v;
        }
        if (m == 0) {  // lane 0 writes rows 0..7, lane 16 writes rows 8..15
            const int pbase = t * 16 + hi * 8;
#pragma unroll
            for (int r = 0; r < 8; ++r) {
                const int pp = pbase + r;
                if (pp < NPAIR) sLogit[pp] = lg[r] + b2v;
            }
        }
        // inter_sum: lane L + lane L+16 cover all 64 K for row m
        isum += __shfl_xor(isum, 16);
        if (hi == 0 && p < NPAIR) sISum[p] = isum;
    }
    __syncthreads();

    // ---- phase 4: block softmax over 1225 pairs + attended sum ---------
    float mx = -3.4e38f;
    for (int p = tid; p < NPAIR; p += 256) mx = fmaxf(mx, sLogit[p]);
    sRed[tid] = mx;
    __syncthreads();
    for (int s = 128; s > 0; s >>= 1) {
        if (tid < s) sRed[tid] = fmaxf(sRed[tid], sRed[tid + s]);
        __syncthreads();
    }
    mx = sRed[0];
    __syncthreads();

    float se = 0.0f, sa = 0.0f;
    for (int p = tid; p < NPAIR; p += 256) {
        const float e = __expf(sLogit[p] - mx);
        se += e;
        sa += e * sISum[p];
    }
    sRed[tid] = se;
    __syncthreads();
    for (int s = 128; s > 0; s >>= 1) {
        if (tid < s) sRed[tid] += sRed[tid + s];
        __syncthreads();
    }
    se = sRed[0];
    __syncthreads();
    sRed[tid] = sa;
    __syncthreads();
    for (int s = 128; s > 0; s >>= 1) {
        if (tid < s) sRed[tid] += sRed[tid + s];
        __syncthreads();
    }
    if (tid == 0)
        out[b] = sLinear + lin_b[0] + sRed[0] / se;
}

extern "C" void kernel_launch(void* const* d_in, const int* in_sizes, int n_in,
                              void* d_out, int out_size, void* d_ws, size_t ws_size,
                              hipStream_t stream) {
    (void)in_sizes; (void)n_in; (void)d_ws; (void)ws_size;
    const int*   x     = (const int*)  d_in[0];
    const float* emb   = (const float*)d_in[1];
    const float* W1    = (const float*)d_in[2];
    const float* b1    = (const float*)d_in[3];
    const float* w2    = (const float*)d_in[4];
    const float* b2    = (const float*)d_in[5];
    const float* lin_w = (const float*)d_in[6];
    const float* lin_b = (const float*)d_in[7];
    float* out = (float*)d_out;
    const int batch = out_size;   // 2048
    afm_wmma_kernel<<<batch, 256, 0, stream>>>(x, emb, W1, b1, w2, b2,
                                               lin_w, lin_b, out);
}
